// ScaledDotProductAttention_50912542327462
// MI455X (gfx1250) — matrix operand order repaired and checked by reading, not compiled, not run on hardware
//
#include <hip/hip_runtime.h>

// Problem constants (reference: B=2, S=4096, E=1024)
#define BB 2
#define SS 4096
#define EE 1024
#define MS (BB * SS)          // 8192 flattened rows
#define SCALE 0.03125f        // 1/sqrt(1024)
#define NINF_F (-1000000000.0f)

typedef __attribute__((ext_vector_type(16))) __bf16 v16bf;
typedef __attribute__((ext_vector_type(8)))  float  v8f;
typedef unsigned int v4u __attribute__((ext_vector_type(4)));
typedef int v8i __attribute__((ext_vector_type(8)));
typedef int v4i __attribute__((ext_vector_type(4)));

union Frag {
  v16bf v;
  uint4 u[2];
  unsigned short s[16];
};

__device__ __forceinline__ unsigned short f32_to_bf16(float f) {
  union { float f; unsigned u; } x;
  x.f = f;
  unsigned r = x.u + 0x7FFFu + ((x.u >> 16) & 1u);  // round-to-nearest-even
  return (unsigned short)(r >> 16);
}

__device__ __forceinline__ v8f wmma_bf16(const Frag& a, const Frag& b, v8f c) {
  return __builtin_amdgcn_wmma_f32_16x16x32_bf16(false, a.v, false, b.v,
                                                 (short)0, c, false, false);
}

// A fragment (16-bit, 16x32): s[0..7]=K k0+8c.., s[8..15]=K k0+16+8c..
__device__ __forceinline__ void load_a(Frag& f, const unsigned short* row,
                                       int k0, int c) {
  f.u[0] = *(const uint4*)(row + k0 + 8 * c);
  f.u[1] = *(const uint4*)(row + k0 + 16 + 8 * c);
}
// B fragment (16-bit, 32x16): s[0..7]=K k0+8c.., s[8..15]=K k0+16+8c..
__device__ __forceinline__ void load_b(Frag& f, const unsigned short* row,
                                       int k0, int c) {
  f.u[0] = *(const uint4*)(row + k0 + 8 * c);
  f.u[1] = *(const uint4*)(row + k0 + 16 + 8 * c);
}

// Generic LDS pointer -> LDS byte offset (flat aperture truncates to addr[31:0]).
__device__ __forceinline__ unsigned lds_off_u32(const void* p) {
  return (unsigned)(unsigned long long)p;
}

// GLOBAL_LOAD_ASYNC_TO_LDS_B128: per-lane 16B global -> LDS, ASYNCcnt-tracked.
__device__ __forceinline__ void async_g2lds_b128(unsigned lds_byte_off,
                                                 const void* gaddr) {
  asm volatile("global_load_async_to_lds_b128 %0, %1, off"
               :: "v"(lds_byte_off), "v"(gaddr)
               : "memory");
}
__device__ __forceinline__ void wait_asynccnt0() {
  asm volatile("s_wait_asynccnt 0x0" ::: "memory");
}

// TDM: load a 2D tile (rows x row_elems of 2-byte elements) global -> LDS.
// D# packed per CDNA5 ISA 8.3/8.4. 6-arg builtin: (g0, g1, g2, g3, g4, cpol).
__device__ __forceinline__ void tdm_load_2d_bf16_to_lds(
    unsigned lds_addr, const void* gptr, unsigned rows, unsigned row_elems,
    unsigned row_stride_elems) {
  const unsigned long long ga = (unsigned long long)gptr;
  v4u g0;
  g0.x = 1u;                                              // count=1, user desc
  g0.y = lds_addr;                                        // lds_addr [63:32]
  g0.z = (unsigned)(ga & 0xFFFFFFFFu);                    // global_addr lo
  g0.w = (unsigned)((ga >> 32) & 0x01FFFFFFu) | (2u << 30);  // addr hi | type=2
  v8i g1;
  g1[0] = (int)(1u << 16);                                // data_size=1 (2B)
  g1[1] = (int)((row_elems & 0xFFFFu) << 16);             // tensor_dim0 lo16
  g1[2] = (int)(((row_elems >> 16) & 0xFFFFu) |           // tensor_dim0 hi16
                ((rows & 0xFFFFu) << 16));                // tensor_dim1 lo16
  g1[3] = (int)(((rows >> 16) & 0xFFFFu) |                // tensor_dim1 hi16
                ((row_elems & 0xFFFFu) << 16));           // tile_dim0
  g1[4] = (int)(rows & 0xFFFFu);                          // tile_dim1 (dim2=0)
  g1[5] = (int)row_stride_elems;                          // tensor_dim0_stride lo
  g1[6] = 0;
  g1[7] = 0;
  const v4i z4 = {0, 0, 0, 0};                            // groups 2/3 unused (2D)
  const v8i z8 = {0, 0, 0, 0, 0, 0, 0, 0};
  __builtin_amdgcn_tensor_load_to_lds(g0, g1, z4, z4, z8, 0);
}

// ---------------------------------------------------------------------------
// K0: element-wise f32 -> bf16 (8 elements/thread, b128 in / b128 out)
// ---------------------------------------------------------------------------
__global__ void __launch_bounds__(256)
cvt_f32_bf16(const float* __restrict__ src, unsigned short* __restrict__ dst,
             int n8) {
  const int i = blockIdx.x * 256 + threadIdx.x;
  if (i >= n8) return;
  const float4* s = (const float4*)(src + (size_t)i * 8);
  const float4 f0 = s[0], f1 = s[1];
  union { uint4 q; unsigned short h[8]; } pk;
  pk.h[0] = f32_to_bf16(f0.x); pk.h[1] = f32_to_bf16(f0.y);
  pk.h[2] = f32_to_bf16(f0.z); pk.h[3] = f32_to_bf16(f0.w);
  pk.h[4] = f32_to_bf16(f1.x); pk.h[5] = f32_to_bf16(f1.y);
  pk.h[6] = f32_to_bf16(f1.z); pk.h[7] = f32_to_bf16(f1.w);
  *(uint4*)(dst + (size_t)i * 8) = pk.q;
}

// ---------------------------------------------------------------------------
// K1: out = X @ W^T + b (nn.Linear), all-bf16 operands, double-buffered.
// Xb [MS,EE] bf16, Wb [EE,EE] bf16, bias f32. Output bf16 row-major
// (transposed==0) or [EE, MS] (transposed==1). 16x64 strip per wave.
// ---------------------------------------------------------------------------
__global__ void __launch_bounds__(256)
proj_gemm_bf16_wmma(const unsigned short* __restrict__ Xb,
                    const unsigned short* __restrict__ Wb,
                    const float* __restrict__ bias,
                    unsigned short* __restrict__ out, int transposed) {
  const int lane = threadIdx.x & 31;
  const int wave = threadIdx.x >> 5;
  const int t  = blockIdx.x * 8 + wave;
  const int m0 = (t >> 4) << 4;   // 16 N-super-tiles of 64 across EE
  const int n0 = (t & 15) << 6;
  const int c   = lane >> 4;
  const int r16 = lane & 15;

  const unsigned short* ax = Xb + (size_t)(m0 + r16) * EE;
  const unsigned short* bw[4];
#pragma unroll
  for (int j = 0; j < 4; ++j) bw[j] = Wb + (size_t)(n0 + 16 * j + r16) * EE;

  v8f acc[4] = {{}, {}, {}, {}};
  Frag a_c, b_c[4];
  load_a(a_c, ax, 0, c);
#pragma unroll
  for (int j = 0; j < 4; ++j) load_b(b_c[j], bw[j], 0, c);

  for (int k0 = 0; k0 < EE; k0 += 32) {
    const int kn = (k0 + 32 < EE) ? k0 + 32 : k0;   // branchless next index
    __builtin_prefetch(ax + k0 + 256, 0, 1);
    Frag a_n, b_n[4];
    load_a(a_n, ax, kn, c);
#pragma unroll
    for (int j = 0; j < 4; ++j) load_b(b_n[j], bw[j], kn, c);
#pragma unroll
    for (int j = 0; j < 4; ++j) acc[j] = wmma_bf16(a_c, b_c[j], acc[j]);
    a_c = a_n;
#pragma unroll
    for (int j = 0; j < 4; ++j) b_c[j] = b_n[j];
  }

#pragma unroll
  for (int j = 0; j < 4; ++j) {
    const int nj = n0 + 16 * j;
    const float bb = bias[nj + r16];
    if (!transposed) {
      unsigned short* po = out + (size_t)(m0 + 8 * c) * EE + nj + r16;
#pragma unroll
      for (int r = 0; r < 8; ++r) po[(size_t)r * EE] = f32_to_bf16(acc[j][r] + bb);
    } else {
      union { uint4 q; unsigned short s[8]; } pk;
#pragma unroll
      for (int r = 0; r < 8; ++r) pk.s[r] = f32_to_bf16(acc[j][r] + bb);
      *(uint4*)(out + (size_t)(nj + r16) * MS + m0 + 8 * c) = pk.q;
    }
  }
}

// ---------------------------------------------------------------------------
// Shared QK^T tile-pair routine for K2 (double-buffered over E).
// qrow points at the LDS-staged Q row, kb0/kb1 at two 16-row K tiles.
// ---------------------------------------------------------------------------
__device__ __forceinline__ void qk_tile_pair(const unsigned short* qrow,
                                             const unsigned short* kb0,
                                             const unsigned short* kb1,
                                             int c, v8f acc[2]) {
  Frag a_c, f0_c, f1_c;
  load_a(a_c, qrow, 0, c);
  load_b(f0_c, kb0, 0, c);
  load_b(f1_c, kb1, 0, c);
  for (int e0 = 0; e0 < EE; e0 += 32) {
    const int en = (e0 + 32 < EE) ? e0 + 32 : e0;
    Frag a_n, f0_n, f1_n;
    load_a(a_n, qrow, en, c);
    load_b(f0_n, kb0, en, c);
    load_b(f1_n, kb1, en, c);
    acc[0] = wmma_bf16(a_c, f0_c, acc[0]);
    acc[1] = wmma_bf16(a_c, f1_c, acc[1]);
    a_c = a_n; f0_c = f0_n; f1_c = f1_n;
  }
}

// ---------------------------------------------------------------------------
// K2: scores = Q K^T * scale, mask, softmax -> P (bf16, [MS, SS]).
// One 16-row Q block per workgroup. Q staged in LDS via the Tensor Data
// Mover (TENSORcnt); the 16 mask rows staged via async b128 (ASYNCcnt).
// Each wave owns a contiguous 512-col kv strip, two score tiles per Q-frag.
// ---------------------------------------------------------------------------
__global__ void __launch_bounds__(256)
attn_scores_softmax_wmma(const unsigned short* __restrict__ Qw,
                         const unsigned short* __restrict__ Kw,
                         const unsigned char* __restrict__ mask,
                         unsigned short* __restrict__ P) {
  __shared__ __align__(16) unsigned short qs[16 * EE];    // 32 KB
  __shared__ __align__(16) unsigned char  msk[16 * SS];   // 64 KB
  __shared__ float pm[8][16];
  __shared__ float pl[8][16];

  const int lane = threadIdx.x & 31;
  const int wave = threadIdx.x >> 5;
  const int b  = blockIdx.x >> 8;
  const int q0 = (blockIdx.x & 255) << 4;
  const int c = lane >> 4;
  const int n = lane & 15;

  // TDM: one wave issues the 16x1024 bf16 tile load for the whole block.
  if (wave == 0)
    tdm_load_2d_bf16_to_lds(lds_off_u32(qs), Qw + ((size_t)b * SS + q0) * EE,
                            16u, (unsigned)EE, (unsigned)EE);

  // Async-copy the q-block's 16 mask rows (16 x 4096 B) into LDS.
  {
    const unsigned char* msrc = mask + (size_t)q0 * SS;
    const unsigned mbase = lds_off_u32(msk);
    for (int i = threadIdx.x; i < (16 * SS) / 16; i += 256)
      async_g2lds_b128(mbase + i * 16, msrc + i * 16);
    wait_asynccnt0();
  }
  __builtin_amdgcn_s_wait_tensorcnt((short)0);
  __syncthreads();

  float m_r[8], l_r[8];
#pragma unroll
  for (int r = 0; r < 8; ++r) { m_r[r] = -__builtin_inff(); l_r[r] = 0.f; }

  const unsigned short* qrow = qs + n * EE;   // A row m = lane&15

  // ---- pass 1: online row max / running sum ----
  for (int it = 0; it < 16; ++it) {
    const int k0 = (wave << 9) + (it << 5);   // wave*512 + it*32
    const unsigned short* kb0 = Kw + ((size_t)b * SS + k0 + n) * EE;
    const unsigned short* kb1 = kb0 + (size_t)16 * EE;
    __builtin_prefetch(kb0 + (size_t)32 * EE, 0, 1);  // next tile pair
    v8f acc[2] = {{}, {}};
    qk_tile_pair(qrow, kb0, kb1, c, acc);
#pragma unroll
    for (int h = 0; h < 2; ++h) {
      const int kt = k0 + 16 * h;
      const unsigned char* mrow = msk + (size_t)(8 * c) * SS + kt + n;
#pragma unroll
      for (int r = 0; r < 8; ++r) {
        float s = acc[h][r] * SCALE;
        if (mrow[(size_t)r * SS]) s = NINF_F;
        float t2 = s;                          // row max across 16 lanes
        t2 = fmaxf(t2, __shfl_xor(t2, 1));
        t2 = fmaxf(t2, __shfl_xor(t2, 2));
        t2 = fmaxf(t2, __shfl_xor(t2, 4));
        t2 = fmaxf(t2, __shfl_xor(t2, 8));
        const float mnew = fmaxf(m_r[r], t2);
        float p = __expf(s - mnew);
        p += __shfl_xor(p, 1);
        p += __shfl_xor(p, 2);
        p += __shfl_xor(p, 4);
        p += __shfl_xor(p, 8);
        l_r[r] = l_r[r] * __expf(m_r[r] - mnew) + p;
        m_r[r] = mnew;
      }
    }
  }

  if (n == 0) {   // lanes 0 (rows 0-7) and 16 (rows 8-15) publish partials
#pragma unroll
    for (int r = 0; r < 8; ++r) {
      pm[wave][8 * c + r] = m_r[r];
      pl[wave][8 * c + r] = l_r[r];
    }
  }
  __syncthreads();

  float mf[8], invl[8];
#pragma unroll
  for (int r = 0; r < 8; ++r) {
    const int row = 8 * c + r;
    float mm = -__builtin_inff();
#pragma unroll
    for (int w = 0; w < 8; ++w) mm = fmaxf(mm, pm[w][row]);
    float ll = 0.f;
#pragma unroll
    for (int w = 0; w < 8; ++w) ll += pl[w][row] * __expf(pm[w][row] - mm);
    mf[r] = mm;
    invl[r] = 1.0f / ll;
  }

  // ---- pass 2: recompute scores, write normalized probabilities ----
  for (int it = 0; it < 16; ++it) {
    const int k0 = (wave << 9) + (it << 5);
    const unsigned short* kb0 = Kw + ((size_t)b * SS + k0 + n) * EE;
    const unsigned short* kb1 = kb0 + (size_t)16 * EE;
    __builtin_prefetch(kb0 + (size_t)32 * EE, 0, 1);
    v8f acc[2] = {{}, {}};
    qk_tile_pair(qrow, kb0, kb1, c, acc);
#pragma unroll
    for (int h = 0; h < 2; ++h) {
      const int kt = k0 + 16 * h;
      const unsigned char* mrow = msk + (size_t)(8 * c) * SS + kt + n;
      unsigned short* prow = P + ((size_t)b * SS + q0 + 8 * c) * SS + kt + n;
#pragma unroll
      for (int r = 0; r < 8; ++r) {
        float s = acc[h][r] * SCALE;
        if (mrow[(size_t)r * SS]) s = NINF_F;
        const float p = __expf(s - mf[r]) * invl[r];
        prow[(size_t)r * SS] = f32_to_bf16(p);
      }
    }
  }
}

// ---------------------------------------------------------------------------
// K3: out = P @ V using VT [EE, MS] bf16 (per-batch columns b*SS + j).
// One 16x64 f32 output strip per wave, double-buffered over the kv axis.
// ---------------------------------------------------------------------------
__global__ void __launch_bounds__(256)
pv_gemm_wmma(const unsigned short* __restrict__ P,
             const unsigned short* __restrict__ VT,
             float* __restrict__ out) {
  const int lane = threadIdx.x & 31;
  const int wave = threadIdx.x >> 5;
  const int t  = blockIdx.x * 8 + wave;
  const int m0 = (t >> 4) << 4;
  const int f0 = (t & 15) << 6;
  const int c = lane >> 4;
  const int n = lane & 15;
  const int b = m0 >> 12;   // batch of this flat row block

  const unsigned short* prow = P + (size_t)(m0 + n) * SS;
  const unsigned short* vrow[4];
#pragma unroll
  for (int j = 0; j < 4; ++j)
    vrow[j] = VT + (size_t)(f0 + 16 * j + n) * MS + (size_t)b * SS;

  v8f acc[4] = {{}, {}, {}, {}};
  Frag a_c, b_c[4];
  load_a(a_c, prow, 0, c);
#pragma unroll
  for (int j = 0; j < 4; ++j) load_b(b_c[j], vrow[j], 0, c);

  for (int j0 = 0; j0 < SS; j0 += 32) {
    const int jn = (j0 + 32 < SS) ? j0 + 32 : j0;
    __builtin_prefetch(prow + j0 + 256, 0, 1);
    Frag a_n, b_n[4];
    load_a(a_n, prow, jn, c);
#pragma unroll
    for (int j = 0; j < 4; ++j) load_b(b_n[j], vrow[j], jn, c);
#pragma unroll
    for (int j = 0; j < 4; ++j) acc[j] = wmma_bf16(a_c, b_c[j], acc[j]);
    a_c = a_n;
#pragma unroll
    for (int j = 0; j < 4; ++j) b_c[j] = b_n[j];
  }
#pragma unroll
  for (int j = 0; j < 4; ++j) {
    float* po = out + (size_t)(m0 + 8 * c) * EE + f0 + 16 * j + n;
#pragma unroll
    for (int r = 0; r < 8; ++r) po[(size_t)r * EE] = acc[j][r];
  }
}

// ---------------------------------------------------------------------------
extern "C" void kernel_launch(void* const* d_in, const int* in_sizes, int n_in,
                              void* d_out, int out_size, void* d_ws, size_t ws_size,
                              hipStream_t stream) {
  (void)in_sizes; (void)n_in; (void)out_size; (void)ws_size;
  const float* query = (const float*)d_in[0];
  const float* key   = (const float*)d_in[1];
  const float* value = (const float*)d_in[2];
  const unsigned char* mask = (const unsigned char*)d_in[3];  // jnp bool = 1 byte
  const float* Wq = (const float*)d_in[4];
  const float* bq = (const float*)d_in[5];
  const float* Wk = (const float*)d_in[6];
  const float* bk = (const float*)d_in[7];
  const float* Wv = (const float*)d_in[8];
  const float* bv = (const float*)d_in[9];

  // Workspace (bytes): Q 16MB | K 16MB | VT 16MB | P 64MB  = 112 MB.
  // The P region is aliased during phase 1 as the bf16 copies of X and W
  // (they are dead before P is first written by K2).
  char* w = (char*)d_ws;
  unsigned short* Qw  = (unsigned short*)(w);
  unsigned short* Kw  = (unsigned short*)(w + (size_t)MS * EE * 2);
  unsigned short* VTw = (unsigned short*)(w + (size_t)MS * EE * 4);
  char* pr = w + (size_t)MS * EE * 6;             // 64 MB phase region
  unsigned short* Pw  = (unsigned short*)pr;
  unsigned short* Xqb = (unsigned short*)(pr);                          // 16MB
  unsigned short* Xkb = (unsigned short*)(pr + (size_t)MS * EE * 2);    // 16MB
  unsigned short* Xvb = (unsigned short*)(pr + (size_t)MS * EE * 4);    // 16MB
  unsigned short* Wqb = (unsigned short*)(pr + (size_t)MS * EE * 6);    // 2MB
  unsigned short* Wkb = Wqb + (size_t)EE * EE;
  unsigned short* Wvb = Wkb + (size_t)EE * EE;

  const dim3 blk(256);
  const int nX8 = MS * EE / 8, nW8 = EE * EE / 8;

  hipLaunchKernelGGL(cvt_f32_bf16, dim3(nX8 / 256), blk, 0, stream, query, Xqb, nX8);
  hipLaunchKernelGGL(cvt_f32_bf16, dim3(nX8 / 256), blk, 0, stream, key,   Xkb, nX8);
  hipLaunchKernelGGL(cvt_f32_bf16, dim3(nX8 / 256), blk, 0, stream, value, Xvb, nX8);
  hipLaunchKernelGGL(cvt_f32_bf16, dim3(nW8 / 256), blk, 0, stream, Wq, Wqb, nW8);
  hipLaunchKernelGGL(cvt_f32_bf16, dim3(nW8 / 256), blk, 0, stream, Wk, Wkb, nW8);
  hipLaunchKernelGGL(cvt_f32_bf16, dim3(nW8 / 256), blk, 0, stream, Wv, Wvb, nW8);

  const int gemmBlocks = (MS / 16) * (EE / 64) / 8;  // 1024
  hipLaunchKernelGGL(proj_gemm_bf16_wmma, dim3(gemmBlocks), blk, 0, stream,
                     Xqb, Wqb, bq, Qw, 0);
  hipLaunchKernelGGL(proj_gemm_bf16_wmma, dim3(gemmBlocks), blk, 0, stream,
                     Xkb, Wkb, bk, Kw, 0);
  hipLaunchKernelGGL(proj_gemm_bf16_wmma, dim3(gemmBlocks), blk, 0, stream,
                     Xvb, Wvb, bv, VTw, 1);
  hipLaunchKernelGGL(attn_scores_softmax_wmma, dim3(BB * SS / 16), blk, 0, stream,
                     Qw, Kw, mask, Pw);
  hipLaunchKernelGGL(pv_gemm_wmma, dim3(gemmBlocks), blk, 0, stream,
                     Pw, VTw, (float*)d_out);
}
